// FP8DiffusionTransformer_32985348833539
// MI455X (gfx1250) — matrix operand order repaired and checked by reading, not compiled, not run on hardware
//
#include <hip/hip_runtime.h>
#include <hip/hip_bf16.h>
#include <math.h>

typedef __attribute__((ext_vector_type(16))) _Float16 v16h;
typedef __attribute__((ext_vector_type(8)))  _Float16 v8h;
typedef __attribute__((ext_vector_type(8)))  float    v8f;

#define DIMC   1024
#define NTOK   321
#define MROWS  1284      // B * NTOK
#define MP16R  1296      // round16(MROWS)
#define NPAD   384       // padded attention seq len (mult of 64 and 32)
#define HDIM   64
#define BH     64        // B * HEADS
#define MLPD   4096
#define MODW   6144      // 6*DIM

// ---------------------------------------------------------------------------
// small elementwise / helper kernels
// ---------------------------------------------------------------------------
__global__ void cvt_kernel(const float* __restrict__ in, _Float16* __restrict__ out, long n) {
    long i = (long)blockIdx.x * blockDim.x + threadIdx.x;
    long stride = (long)gridDim.x * blockDim.x;
    for (; i < n; i += stride) out[i] = (_Float16)in[i];
}

__global__ void embed_text_kernel(const float* __restrict__ W_tok, const int* __restrict__ tokens,
                                  const float* __restrict__ pos_text, const float* __restrict__ cls,
                                  float* __restrict__ X) {
    int r = blockIdx.x;              // 0 .. B*65-1
    int b = r / 65, n = r % 65;
    float* xr = X + ((long)(b * NTOK + n)) * DIMC;
    if (n < 64) {
        int t = tokens[b * 64 + n];
        for (int d = threadIdx.x; d < DIMC; d += blockDim.x)
            xr[d] = W_tok[(long)d * 50000 + t] + pos_text[(long)n * DIMC + d];
    } else {
        for (int d = threadIdx.x; d < DIMC; d += blockDim.x) xr[d] = cls[d];
    }
}

__global__ void patch_kernel(const float* __restrict__ img, _Float16* __restrict__ Pm) {
    long idx = (long)blockIdx.x * blockDim.x + threadIdx.x;   // 1024*768 total
    if (idx >= 1024l * 768) return;
    int col = (int)(idx % 768);
    int row = (int)(idx / 768);
    int b = row >> 8, p = row & 255, pr = p >> 4, pc = p & 15;
    int ch = col >> 8, q = col & 255, py = q >> 4, px = q & 15;
    Pm[idx] = (_Float16)img[(((long)(b * 3 + ch)) * 256 + pr * 16 + py) * 256 + pc * 16 + px];
}

__global__ void temb_kernel(const int* __restrict__ ts, float* __restrict__ raw) {
    int b = blockIdx.x;              // 4
    int i = threadIdx.x;             // 128
    float fr = expf(-9.210340372f * (float)i / 128.f);
    float a = (float)ts[b] * fr;
    raw[b * 256 + i]       = cosf(a);
    raw[b * 256 + 128 + i] = sinf(a);
}

__global__ void silu_kernel(const float* __restrict__ in, float* __restrict__ out, int n) {
    int i = blockIdx.x * blockDim.x + threadIdx.x;
    if (i < n) { float x = in[i]; out[i] = x / (1.f + expf(-x)); }
}

template<bool SILU_OUT>
__global__ __launch_bounds__(256) void small_linear(const float* __restrict__ in,
                                                    const float* __restrict__ W,
                                                    const float* __restrict__ bias,
                                                    float* __restrict__ out,
                                                    int Bn, int K, int Nout) {
    int gid = blockIdx.x * (blockDim.x >> 5) + (threadIdx.x >> 5);
    if (gid >= Bn * Nout) return;
    int b = gid / Nout, o = gid % Nout;
    int lane = threadIdx.x & 31;
    float s = 0.f;
    const float* ir = in + (long)b * K;
    const float* wr = W + (long)o * K;
    for (int k = lane; k < K; k += 32) s += ir[k] * wr[k];
    for (int m = 16; m; m >>= 1) s += __shfl_xor(s, m, 32);
    if (lane == 0) {
        float r = s + (bias ? bias[o] : 0.f);
        if (SILU_OUT) r = r / (1.f + expf(-r));
        out[(long)b * Nout + o] = r;
    }
}

// LayerNorm (+ optional adaLN modulation), fp32 in -> fp16 out
__global__ __launch_bounds__(256) void ln_mod_kernel(const float* __restrict__ X,
                                                     const float* __restrict__ g,
                                                     const float* __restrict__ bt,
                                                     const float* __restrict__ mod,
                                                     int sh_off, int sc_off,
                                                     _Float16* __restrict__ out,
                                                     int tpb, int row_base) {
    int r = blockIdx.x;
    int bidx = r / tpb, n = r % tpb;
    const float* xr = X + ((long)(bidx * NTOK + row_base + n)) * DIMC;
    __shared__ float sred[8];
    float s = 0.f;
    for (int i = threadIdx.x; i < DIMC; i += 256) s += xr[i];
    for (int m = 16; m; m >>= 1) s += __shfl_xor(s, m, 32);
    if ((threadIdx.x & 31) == 0) sred[threadIdx.x >> 5] = s;
    __syncthreads();
    float mean = 0.f;
    for (int w = 0; w < 8; w++) mean += sred[w];
    mean *= (1.f / DIMC);
    __syncthreads();
    float v = 0.f;
    for (int i = threadIdx.x; i < DIMC; i += 256) { float d = xr[i] - mean; v += d * d; }
    for (int m = 16; m; m >>= 1) v += __shfl_xor(v, m, 32);
    if ((threadIdx.x & 31) == 0) sred[threadIdx.x >> 5] = v;
    __syncthreads();
    float var = 0.f;
    for (int w = 0; w < 8; w++) var += sred[w];
    float rstd = rsqrtf(var * (1.f / DIMC) + 1e-5f);
    const float* mrow = mod ? (mod + (long)bidx * MODW) : nullptr;
    for (int i = threadIdx.x; i < DIMC; i += 256) {
        float y = (xr[i] - mean) * rstd * g[i] + bt[i];
        if (mrow) y = y * (1.f + mrow[sc_off + i]) + mrow[sh_off + i];
        out[(long)r * DIMC + i] = (_Float16)y;
    }
}

// RoPE on q,k + transpose v -> Vt, fp16 outputs padded/zero-filled to NPAD
__global__ __launch_bounds__(64) void rope_kernel(const float* __restrict__ qf,
                                                  const float* __restrict__ kf,
                                                  const float* __restrict__ vf,
                                                  _Float16* __restrict__ Qh,
                                                  _Float16* __restrict__ Kh,
                                                  _Float16* __restrict__ Vt) {
    int z = blockIdx.x;        // b*16+h
    int n = blockIdx.y;        // 0..NPAD-1
    int d = threadIdx.x;       // 0..63
    int b = z >> 4, h = z & 15;
    long qo = ((long)z * NPAD + n) * HDIM + d;
    long vo = ((long)z * HDIM + d) * NPAD + n;
    if (n < NTOK) {
        long row = ((long)(b * NTOK + n)) * DIMC + h * HDIM;
        float q = qf[row + d], k = kf[row + d], v = vf[row + d];
        int j = (d < 32) ? d : d - 32;
        float inv = expf(-9.210340372f * (2.f * (float)j) / 64.f);
        float ang = (float)n * inv;
        float c = cosf(ang), s = sinf(ang);
        float qr = (d < 32) ? -qf[row + 2 * d + 1] : qf[row + 2 * (d - 32)];
        float kr = (d < 32) ? -kf[row + 2 * d + 1] : kf[row + 2 * (d - 32)];
        Qh[qo] = (_Float16)(q * c + qr * s);
        Kh[qo] = (_Float16)(k * c + kr * s);
        Vt[vo] = (_Float16)v;
    } else {
        Qh[qo] = (_Float16)0.f; Kh[qo] = (_Float16)0.f; Vt[vo] = (_Float16)0.f;
    }
}

// softmax over cols [0,321) with scale 1/8; zero cols [321,NPAD)
__global__ __launch_bounds__(128) void softmax_kernel(float* __restrict__ S) {
    int z = blockIdx.y, n = blockIdx.x;     // n < 321
    float* row = S + ((long)z * NPAD + n) * NPAD;
    __shared__ float sred[4];
    float mx = -1e30f;
    for (int j = threadIdx.x; j < NTOK; j += 128) mx = fmaxf(mx, row[j] * 0.125f);
    for (int m = 16; m; m >>= 1) mx = fmaxf(mx, __shfl_xor(mx, m, 32));
    if ((threadIdx.x & 31) == 0) sred[threadIdx.x >> 5] = mx;
    __syncthreads();
    float M = fmaxf(fmaxf(sred[0], sred[1]), fmaxf(sred[2], sred[3]));
    __syncthreads();
    float sum = 0.f;
    for (int j = threadIdx.x; j < NTOK; j += 128) sum += expf(row[j] * 0.125f - M);
    for (int m = 16; m; m >>= 1) sum += __shfl_xor(sum, m, 32);
    if ((threadIdx.x & 31) == 0) sred[threadIdx.x >> 5] = sum;
    __syncthreads();
    float tot = sred[0] + sred[1] + sred[2] + sred[3];
    float inv = 1.f / tot;
    for (int j = threadIdx.x; j < NPAD; j += 128)
        row[j] = (j < NTOK) ? expf(row[j] * 0.125f - M) * inv : 0.f;
}

// ---------------------------------------------------------------------------
// Generic batched NT WMMA GEMM: C[z][m][n] = sum_k A[z][m][k] * B[z][n][k]
// Wave computes a 16x64 tile of C; block = 4 waves (64 rows x 64 cols).
// k-loop is software-pipelined: fragments for k+32 are issued before the
// WMMAs consuming the k fragments, so the matrix pipe overlaps the loads.
// ---------------------------------------------------------------------------
struct EpiStoreF32 {
    float* C; int ldc; long sCz; int Mlim;
    __device__ void operator()(int z, int m0, int n0, int lane, const v8f& c) const {
        int col = n0 + (lane & 15), g = lane >> 4;
        #pragma unroll
        for (int r = 0; r < 8; r++) {
            int row = m0 + r + 8 * g;
            if (row < Mlim) C[(long)z * sCz + (long)row * ldc + col] = c[r];
        }
    }
};
struct EpiPatch {   // patch-embed: += pos_img, write into x rows 65+
    float* X; const float* pos;
    __device__ void operator()(int z, int m0, int n0, int lane, const v8f& c) const {
        int col = n0 + (lane & 15), g = lane >> 4;
        #pragma unroll
        for (int r = 0; r < 8; r++) {
            int m = m0 + r + 8 * g;                 // b*256+p, M=1024 exact
            int b = m >> 8, p = m & 255;
            X[((long)(b * NTOK + 65 + p)) * DIMC + col] = c[r] + pos[(long)p * DIMC + col];
        }
    }
};
struct EpiPV {      // attention output -> o_h fp16 [MP16R][DIM]
    _Float16* O;
    __device__ void operator()(int z, int m0, int n0, int lane, const v8f& c) const {
        int d = n0 + (lane & 15), g = lane >> 4;
        int b = z >> 4, h = z & 15;
        #pragma unroll
        for (int r = 0; r < 8; r++) {
            int n = m0 + r + 8 * g;
            if (n < NTOK) O[((long)(b * NTOK + n)) * DIMC + h * HDIM + d] = (_Float16)c[r];
        }
    }
};
struct EpiResidGate {  // x += gate * C
    float* X; const float* mod; int goff;
    __device__ void operator()(int z, int m0, int n0, int lane, const v8f& c) const {
        int col = n0 + (lane & 15), g = lane >> 4;
        #pragma unroll
        for (int r = 0; r < 8; r++) {
            int row = m0 + r + 8 * g;
            if (row < MROWS) {
                int b = row / NTOK;
                X[(long)row * DIMC + col] += mod[(long)b * MODW + goff + col] * c[r];
            }
        }
    }
};
struct EpiGelu {    // exact-erf gelu -> fp16 h1
    _Float16* H;
    __device__ void operator()(int z, int m0, int n0, int lane, const v8f& c) const {
        int col = n0 + (lane & 15), g = lane >> 4;
        #pragma unroll
        for (int r = 0; r < 8; r++) {
            int row = m0 + r + 8 * g;
            float v = c[r];
            H[(long)row * MLPD + col] = (_Float16)(0.5f * v * (1.f + erff(v * 0.70710678f)));
        }
    }
};
struct EpiFinal {   // + b_final, unpatchify into d_out [B,3,256,256]
    float* out; const float* bf;
    __device__ void operator()(int z, int m0, int n0, int lane, const v8f& c) const {
        int o = n0 + (lane & 15), g = lane >> 4;
        int ch = o >> 8, q = o & 255, py = q >> 4, px = q & 15;
        #pragma unroll
        for (int r = 0; r < 8; r++) {
            int m = m0 + r + 8 * g;                 // b*256+p, M=1024 exact
            int b = m >> 8, p = m & 255, pr = p >> 4, pc = p & 15;
            out[(((long)(b * 3 + ch)) * 256 + pr * 16 + py) * 256 + pc * 16 + px] = c[r] + bf[o];
        }
    }
};

template<bool AF32, class Epi>
__global__ __launch_bounds__(128) void gemm_nt(const void* __restrict__ Av, long sAz, int lda,
                                               const _Float16* __restrict__ Bw, long sBz, int ldb,
                                               int Mp16, int K, Epi epi) {
    int lane = threadIdx.x & 31;
    int wv = threadIdx.x >> 5;
    int m0 = (blockIdx.y * 4 + wv) * 16;
    if (m0 >= Mp16) return;
    int n0 = blockIdx.x * 64;
    int z = blockIdx.z;
    int mm = lane & 15, g = lane >> 4;

    const float* ApF = nullptr;
    const _Float16* ApH = nullptr;
    if constexpr (AF32)
        ApF = (const float*)Av + (long)z * sAz + (long)(m0 + mm) * lda + g * 8;
    else
        ApH = (const _Float16*)Av + (long)z * sAz + (long)(m0 + mm) * lda + g * 8;
    const _Float16* Bp = Bw + (long)z * sBz + (long)(n0 + mm) * ldb + g * 8;

    auto loadA = [&](int k0) -> v16h {
        v16h a;
        if constexpr (AF32) {
            const float* p = ApF + k0;
            #pragma unroll
            for (int i = 0; i < 8; i++) { a[i] = (_Float16)p[i]; a[8 + i] = (_Float16)p[16 + i]; }
        } else {
            const _Float16* p = ApH + k0;
            v8h lo = *(const v8h*)p;
            v8h hi = *(const v8h*)(p + 16);
            #pragma unroll
            for (int i = 0; i < 8; i++) { a[i] = lo[i]; a[8 + i] = hi[i]; }
        }
        return a;
    };
    auto loadB = [&](int k0, int t) -> v16h {
        const _Float16* p = Bp + (long)(t * 16) * ldb + k0;
        v8h b0 = *(const v8h*)p;
        v8h b1 = *(const v8h*)(p + 16);
        v16h b;
        #pragma unroll
        for (int i = 0; i < 8; i++) { b[i] = b0[i]; b[8 + i] = b1[i]; }
        return b;
    };

    v8f acc[4] = {v8f{}, v8f{}, v8f{}, v8f{}};
    // prologue: fragments for k=0
    v16h a_c  = loadA(0);
    v16h b_c0 = loadB(0, 0), b_c1 = loadB(0, 1), b_c2 = loadB(0, 2), b_c3 = loadB(0, 3);
    for (int k0 = 32; k0 < K; k0 += 32) {
        // issue next-step loads before consuming current fragments
        v16h a_n  = loadA(k0);
        v16h b_n0 = loadB(k0, 0), b_n1 = loadB(k0, 1), b_n2 = loadB(k0, 2), b_n3 = loadB(k0, 3);
        acc[0] = __builtin_amdgcn_wmma_f32_16x16x32_f16(false, a_c, false, b_c0, (short)0, acc[0], false, false);
        acc[1] = __builtin_amdgcn_wmma_f32_16x16x32_f16(false, a_c, false, b_c1, (short)0, acc[1], false, false);
        acc[2] = __builtin_amdgcn_wmma_f32_16x16x32_f16(false, a_c, false, b_c2, (short)0, acc[2], false, false);
        acc[3] = __builtin_amdgcn_wmma_f32_16x16x32_f16(false, a_c, false, b_c3, (short)0, acc[3], false, false);
        a_c = a_n; b_c0 = b_n0; b_c1 = b_n1; b_c2 = b_n2; b_c3 = b_n3;
    }
    acc[0] = __builtin_amdgcn_wmma_f32_16x16x32_f16(false, a_c, false, b_c0, (short)0, acc[0], false, false);
    acc[1] = __builtin_amdgcn_wmma_f32_16x16x32_f16(false, a_c, false, b_c1, (short)0, acc[1], false, false);
    acc[2] = __builtin_amdgcn_wmma_f32_16x16x32_f16(false, a_c, false, b_c2, (short)0, acc[2], false, false);
    acc[3] = __builtin_amdgcn_wmma_f32_16x16x32_f16(false, a_c, false, b_c3, (short)0, acc[3], false, false);

    epi(z, m0, n0,      lane, acc[0]);
    epi(z, m0, n0 + 16, lane, acc[1]);
    epi(z, m0, n0 + 32, lane, acc[2]);
    epi(z, m0, n0 + 48, lane, acc[3]);
}

// ---------------------------------------------------------------------------
// host driver
// ---------------------------------------------------------------------------
extern "C" void kernel_launch(void* const* d_in, const int* in_sizes, int n_in,
                              void* d_out, int out_size, void* d_ws, size_t ws_size,
                              hipStream_t stream) {
    (void)in_sizes; (void)n_in; (void)out_size; (void)ws_size;
    const float* noisy    = (const float*)d_in[0];
    const int*   tokens   = (const int*)  d_in[1];
    const int*   tsteps   = (const int*)  d_in[2];
    const float* W_tok    = (const float*)d_in[3];
    const float* pos_text = (const float*)d_in[4];
    const float* W_patch  = (const float*)d_in[5];
    const float* pos_img  = (const float*)d_in[6];
    const float* cls_tok  = (const float*)d_in[7];
    const float* W_t1     = (const float*)d_in[8];
    const float* b_t1     = (const float*)d_in[9];
    const float* W_t2     = (const float*)d_in[10];
    const float* b_t2     = (const float*)d_in[11];
    const float* ln1_g    = (const float*)d_in[12];
    const float* ln1_b    = (const float*)d_in[13];
    const float* Wq       = (const float*)d_in[14];
    const float* Wk       = (const float*)d_in[15];
    const float* Wv       = (const float*)d_in[16];
    const float* Wo       = (const float*)d_in[17];
    const float* ln2_g    = (const float*)d_in[18];
    const float* ln2_b    = (const float*)d_in[19];
    const float* Wm1      = (const float*)d_in[20];
    const float* Wm2      = (const float*)d_in[21];
    const float* Wada     = (const float*)d_in[22];
    const float* b_ada    = (const float*)d_in[23];
    const float* fn_g     = (const float*)d_in[24];
    const float* fn_b     = (const float*)d_in[25];
    const float* W_final  = (const float*)d_in[26];
    const float* b_final  = (const float*)d_in[27];
    float* out = (float*)d_out;

    char* base = (char*)d_ws;
    size_t off = 0;
    auto carve = [&](size_t bytes) -> char* {
        char* p = base + off;
        off = (off + bytes + 255) & ~(size_t)255;
        return p;
    };
    _Float16* Wq_h  = (_Float16*)carve(2l * DIMC * DIMC * 2);
    _Float16* Wk_h  = (_Float16*)carve(2l * DIMC * DIMC * 2);
    _Float16* Wv_h  = (_Float16*)carve(2l * DIMC * DIMC * 2);
    _Float16* Wo_h  = (_Float16*)carve(2l * DIMC * DIMC * 2);
    _Float16* Wm1_h = (_Float16*)carve(2l * MLPD * DIMC * 2);
    _Float16* Wm2_h = (_Float16*)carve(2l * DIMC * MLPD * 2);
    _Float16* Wp_h  = (_Float16*)carve(1024l * 768 * 2);
    _Float16* Wf_h  = (_Float16*)carve(768l * DIMC * 2);
    float*    X     = (float*)   carve((long)MP16R * DIMC * 4);
    _Float16* xh    = (_Float16*)carve((long)MP16R * DIMC * 2);
    _Float16* Pm    = (_Float16*)carve(1024l * 768 * 2);
    float*    qf    = (float*)   carve((long)MROWS * DIMC * 4);
    float*    kf    = (float*)   carve((long)MROWS * DIMC * 4);
    float*    vf    = (float*)   carve((long)MROWS * DIMC * 4);
    _Float16* Qh    = (_Float16*)carve((long)BH * NPAD * HDIM * 2);
    _Float16* Kh    = (_Float16*)carve((long)BH * NPAD * HDIM * 2);
    _Float16* Vt    = (_Float16*)carve((long)BH * HDIM * NPAD * 2);
    float*    S     = (float*)   carve((long)BH * NPAD * NPAD * 4);
    _Float16* o_h   = (_Float16*)carve((long)MP16R * DIMC * 2);
    _Float16* h1    = (_Float16*)carve((long)MP16R * MLPD * 2);
    float*    raw   = (float*)   carve(4l * 256 * 4);
    float*    t1    = (float*)   carve(4l * DIMC * 4);
    float*    temb  = (float*)   carve(4l * DIMC * 4);
    float*    st    = (float*)   carve(4l * DIMC * 4);
    float*    modb  = (float*)   carve(4l * MODW * 4);

    // --- convert GEMM weights to fp16 ---
    auto cvt = [&](const float* s, _Float16* d, long n) {
        cvt_kernel<<<dim3(2048), 256, 0, stream>>>(s, d, n);
    };
    cvt(Wq, Wq_h, 2l * DIMC * DIMC);
    cvt(Wk, Wk_h, 2l * DIMC * DIMC);
    cvt(Wv, Wv_h, 2l * DIMC * DIMC);
    cvt(Wo, Wo_h, 2l * DIMC * DIMC);
    cvt(Wm1, Wm1_h, 2l * MLPD * DIMC);
    cvt(Wm2, Wm2_h, 2l * DIMC * MLPD);
    cvt(W_patch, Wp_h, 1024l * 768);
    cvt(W_final, Wf_h, 768l * DIMC);

    // --- embeddings ---
    embed_text_kernel<<<dim3(4 * 65), 256, 0, stream>>>(W_tok, tokens, pos_text, cls_tok, X);
    patch_kernel<<<dim3((1024 * 768 + 255) / 256), 256, 0, stream>>>(noisy, Pm);
    gemm_nt<false><<<dim3(16, 16, 1), 128, 0, stream>>>(
        (const void*)Pm, 0l, 768, Wp_h, 0l, 768, 1024, 768, EpiPatch{X, pos_img});

    // --- time embedding: raw -> silu(W_t1) -> W_t2 -> silu ---
    temb_kernel<<<dim3(4), 128, 0, stream>>>(tsteps, raw);
    small_linear<true ><<<dim3((4 * DIMC + 7) / 8), 256, 0, stream>>>(raw, W_t1, b_t1, t1, 4, 256, DIMC);
    small_linear<false><<<dim3((4 * DIMC + 7) / 8), 256, 0, stream>>>(t1, W_t2, b_t2, temb, 4, DIMC, DIMC);
    silu_kernel<<<dim3(16), 256, 0, stream>>>(temb, st, 4 * DIMC);

    const dim3 gemm_blk(128);
    const dim3 grid_x1(16, 21, 1);     // N=1024, Mp16=1296
    for (int l = 0; l < 2; l++) {
        long wofs = (long)l * DIMC * DIMC;
        // adaLN modulation: mod = silu(t_emb) @ Wada[l]^T + b_ada[l]
        small_linear<false><<<dim3((4 * MODW + 7) / 8), 256, 0, stream>>>(
            st, Wada + (long)l * MODW * DIMC, b_ada + (long)l * MODW, modb, 4, DIMC, MODW);
        // xn = ln1(x)*(1+sc_msa)+sh_msa
        ln_mod_kernel<<<dim3(MROWS), 256, 0, stream>>>(
            X, ln1_g + l * DIMC, ln1_b + l * DIMC, modb, 0, DIMC, xh, NTOK, 0);
        // q,k,v projections
        gemm_nt<false><<<grid_x1, gemm_blk, 0, stream>>>(
            (const void*)xh, 0l, DIMC, Wq_h + wofs, 0l, DIMC, MP16R, DIMC,
            EpiStoreF32{qf, DIMC, 0l, MROWS});
        gemm_nt<false><<<grid_x1, gemm_blk, 0, stream>>>(
            (const void*)xh, 0l, DIMC, Wk_h + wofs, 0l, DIMC, MP16R, DIMC,
            EpiStoreF32{kf, DIMC, 0l, MROWS});
        gemm_nt<false><<<grid_x1, gemm_blk, 0, stream>>>(
            (const void*)xh, 0l, DIMC, Wv_h + wofs, 0l, DIMC, MP16R, DIMC,
            EpiStoreF32{vf, DIMC, 0l, MROWS});
        // rope + v transpose, fp16, zero-padded
        rope_kernel<<<dim3(BH, NPAD), 64, 0, stream>>>(qf, kf, vf, Qh, Kh, Vt);
        // S = Q K^T  (batched over z = b*16+h)
        gemm_nt<false><<<dim3(NPAD / 64, NPAD / 64, BH), gemm_blk, 0, stream>>>(
            (const void*)Qh, (long)NPAD * HDIM, HDIM, Kh, (long)NPAD * HDIM, HDIM, NPAD, HDIM,
            EpiStoreF32{S, NPAD, (long)NPAD * NPAD, NPAD});
        // softmax (scale folded, zero-pad tail)
        softmax_kernel<<<dim3(NTOK, BH), 128, 0, stream>>>(S);
        // O = P V  (A is fp32 P, converted on the fly)
        gemm_nt<true><<<dim3(1, NPAD / 64, BH), gemm_blk, 0, stream>>>(
            (const void*)S, (long)NPAD * NPAD, NPAD, Vt, (long)HDIM * NPAD, NPAD, NPAD, NPAD,
            EpiPV{o_h});
        // x += g_msa * (o @ Wo^T)
        gemm_nt<false><<<grid_x1, gemm_blk, 0, stream>>>(
            (const void*)o_h, 0l, DIMC, Wo_h + wofs, 0l, DIMC, MP16R, DIMC,
            EpiResidGate{X, modb, 2 * DIMC});
        // xn2 = ln2(x)*(1+sc_mlp)+sh_mlp
        ln_mod_kernel<<<dim3(MROWS), 256, 0, stream>>>(
            X, ln2_g + l * DIMC, ln2_b + l * DIMC, modb, 3 * DIMC, 4 * DIMC, xh, NTOK, 0);
        // h1 = gelu(xn2 @ Wm1^T)
        gemm_nt<false><<<dim3(MLPD / 64, 21, 1), gemm_blk, 0, stream>>>(
            (const void*)xh, 0l, DIMC, Wm1_h + (long)l * MLPD * DIMC, 0l, DIMC, MP16R, DIMC,
            EpiGelu{h1});
        // x += g_mlp * (h1 @ Wm2^T)
        gemm_nt<false><<<grid_x1, gemm_blk, 0, stream>>>(
            (const void*)h1, 0l, MLPD, Wm2_h + (long)l * DIMC * MLPD, 0l, MLPD, MP16R, MLPD,
            EpiResidGate{X, modb, 5 * DIMC});
    }

    // --- final LN over image tokens (rows b*321+65+p) ---
    ln_mod_kernel<<<dim3(1024), 256, 0, stream>>>(
        X, fn_g, fn_b, (const float*)nullptr, 0, 0, xh, 256, 65);
    // out = ln @ W_final^T + b_final, unpatchified
    gemm_nt<false><<<dim3(768 / 64, 16, 1), gemm_blk, 0, stream>>>(
        (const void*)xh, 0l, DIMC, Wf_h, 0l, DIMC, 1024, DIMC,
        EpiFinal{out, b_final});
}
